// DockBase_34591666602701
// MI455X (gfx1250) — matrix operand order repaired and checked by reading, not compiled, not run on hardware
//
#include <hip/hip_runtime.h>

#define NN 50000
#define NE 500000
#define HD 128

typedef __attribute__((ext_vector_type(16))) __bf16 v16bf;
typedef __attribute__((ext_vector_type(8)))  float  v8f;

union BF16x16 { unsigned short u[16]; v16bf v; };

__device__ __forceinline__ unsigned short f2bf(float f) {
  unsigned int u = __float_as_uint(f);
  u += 0x7FFFu + ((u >> 16) & 1u);          // round-to-nearest-even
  return (unsigned short)(u >> 16);
}

__device__ __forceinline__ v8f wmma_bf16(v16bf a, v16bf b, v8f c) {
  return __builtin_amdgcn_wmma_f32_16x16x32_bf16(false, a, false, b, (short)0, c,
                                                 false, false);
}

#define LDS_FENCE() asm volatile("s_wait_dscnt 0" ::: "memory")

// A fragment: 16xK row-major bf16 tile in LDS, K-block of 32 at k0.
// CDNA5 layout: lanes 0-15 row M=lane hold K=k0..k0+7 then k0+16..k0+23;
// lanes 16-31 same rows hold K=k0+8..k0+15 then k0+24..k0+31.
__device__ __forceinline__ v16bf a_frag(const unsigned short* A, int ldK, int k0, int lane) {
  int r = lane & 15, hb = lane >> 4;
  const unsigned short* p = A + r * ldK + k0 + hb * 8;
  BF16x16 f;
#pragma unroll
  for (int j = 0; j < 8; ++j) f.u[j] = p[j];
#pragma unroll
  for (int j = 0; j < 8; ++j) f.u[8 + j] = p[16 + j];
  return f.v;
}

// B fragment: weights stored transposed WT[n][k] bf16 in global memory.
// Lane holds column n = n0 + (lane&15); lanes<16: K=k0..k0+7 then k0+16..k0+23, lanes>=16: K=k0+8..k0+15 then k0+24..k0+31.
__device__ __forceinline__ v16bf b_frag(const unsigned short* WT, int K, int n0, int k0, int lane) {
  int n = n0 + (lane & 15), hb = lane >> 4;
  const unsigned short* p = WT + (size_t)n * K + k0 + hb * 8;
  BF16x16 f;
#pragma unroll
  for (int j = 0; j < 16; ++j) f.u[j] = p[j < 8 ? j : j + 8];
  return f.v;
}

// ---------------- helper kernels ----------------

__global__ void egnn_deg_kernel(const int* dst, float* deg, int E) {
  int i = blockIdx.x * blockDim.x + threadIdx.x;
  if (i < E) atomicAdd(&deg[dst[i]], 1.0f);
}

// W [in][out] fp32 -> WT [out][in] bf16
__global__ void egnn_wconv_kernel(const float* W, unsigned short* WT, int in_dim, int out_dim) {
  int i = blockIdx.x * blockDim.x + threadIdx.x;
  int total = in_dim * out_dim;
  if (i >= total) return;
  int o = i / in_dim, k = i - o * in_dim;
  WT[(size_t)o * in_dim + k] = f2bf(W[(size_t)k * out_dim + o]);
}

// ---------------- edge kernel: one wave per 16 edges ----------------

__global__ __launch_bounds__(64) void egnn_edge_kernel(
    const float* __restrict__ x, const float* __restrict__ eattr,
    const int* __restrict__ ei, const float* __restrict__ pos,
    const unsigned short* __restrict__ mW1T, const float* __restrict__ mb1,
    const unsigned short* __restrict__ mW2T, const float* __restrict__ mb2,
    const unsigned short* __restrict__ aW1T, const float* __restrict__ ab1,
    const float* __restrict__ aW2, const float* __restrict__ ab2,
    float* __restrict__ pos_acc, float* __restrict__ intra, int E) {
  __shared__ unsigned short sA[2][16 * 384];   // staged cat(x_dst,x_src,eattr) bf16
  __shared__ unsigned short sH[2][16 * 128];   // hidden / reused
  __shared__ float sW[2][16];

  int wave = threadIdx.x >> 5, lane = threadIdx.x & 31;
  int tile = blockIdx.x * 2 + wave;
  if (tile * 16 >= E) return;
  int e0 = tile * 16;
  const int* src = ei;
  const int* dst = ei + E;
  unsigned short* A = sA[wave];
  unsigned short* Hb = sH[wave];
  int hb = lane >> 4, colb = lane & 15;

  // ---- stage A tile: 16 rows x 384 cols, fp32 -> bf16 ----
  for (int r = 0; r < 16; ++r) {
    int e = e0 + r;
    int vd = dst[e], vs = src[e];
    float4 fd = *(const float4*)(x + (size_t)vd * HD + lane * 4);
    float4 fs = *(const float4*)(x + (size_t)vs * HD + lane * 4);
    float4 fe = *(const float4*)(eattr + (size_t)e * HD + lane * 4);
    int b0 = r * 384 + lane * 4;
    A[b0 + 0] = f2bf(fd.x); A[b0 + 1] = f2bf(fd.y); A[b0 + 2] = f2bf(fd.z); A[b0 + 3] = f2bf(fd.w);
    A[b0 + 128 + 0] = f2bf(fs.x); A[b0 + 128 + 1] = f2bf(fs.y); A[b0 + 128 + 2] = f2bf(fs.z); A[b0 + 128 + 3] = f2bf(fs.w);
    A[b0 + 256 + 0] = f2bf(fe.x); A[b0 + 256 + 1] = f2bf(fe.y); A[b0 + 256 + 2] = f2bf(fe.z); A[b0 + 256 + 3] = f2bf(fe.w);
  }
  LDS_FENCE();

  // ---- GEMM1: hidden = relu([16,384] @ mW1 + mb1) -> bf16 LDS tile [16,128] ----
#pragma unroll
  for (int nt = 0; nt < 8; ++nt) {
    float bias = mb1[nt * 16 + colb];
    v8f c;
#pragma unroll
    for (int g = 0; g < 8; ++g) c[g] = bias;
#pragma unroll
    for (int kb = 0; kb < 12; ++kb)
      c = wmma_bf16(a_frag(A, 384, kb * 32, lane), b_frag(mW1T, 384, nt * 16, kb * 32, lane), c);
#pragma unroll
    for (int g = 0; g < 8; ++g) {
      int row = g + hb * 8;
      Hb[row * 128 + nt * 16 + colb] = f2bf(fmaxf(c[g], 0.0f));
    }
  }
  LDS_FENCE();

  // ---- GEMM2: m = hidden @ mW2 + mb2 ; keep f32 in regs + bf16 tile (reuse A) ----
  unsigned short* Mb = A;  // A tile dead; reuse first 4KB for m bf16
  v8f mreg[8];
#pragma unroll
  for (int nt = 0; nt < 8; ++nt) {
    float bias = mb2[nt * 16 + colb];
    v8f c;
#pragma unroll
    for (int g = 0; g < 8; ++g) c[g] = bias;
#pragma unroll
    for (int kb = 0; kb < 4; ++kb)
      c = wmma_bf16(a_frag(Hb, 128, kb * 32, lane), b_frag(mW2T, 128, nt * 16, kb * 32, lane), c);
    mreg[nt] = c;
#pragma unroll
    for (int g = 0; g < 8; ++g) {
      int row = g + hb * 8;
      Mb[row * 128 + nt * 16 + colb] = f2bf(c[g]);
    }
  }
  LDS_FENCE();

  // ---- acc MLP: t = relu(m @ aW1 + ab1) [16,256]; w = t @ aW2 + ab2 [16,1] ----
  float wsum[8];
#pragma unroll
  for (int g = 0; g < 8; ++g) wsum[g] = 0.0f;
#pragma unroll
  for (int nt = 0; nt < 16; ++nt) {
    float bias = ab1[nt * 16 + colb];
    v8f c;
#pragma unroll
    for (int g = 0; g < 8; ++g) c[g] = bias;
#pragma unroll
    for (int kb = 0; kb < 4; ++kb)
      c = wmma_bf16(a_frag(Mb, 128, kb * 32, lane), b_frag(aW1T, 128, nt * 16, kb * 32, lane), c);
    float a2 = aW2[nt * 16 + colb];
#pragma unroll
    for (int g = 0; g < 8; ++g) wsum[g] += fmaxf(c[g], 0.0f) * a2;
  }
  // reduce over 16 columns within each half-wave
#pragma unroll
  for (int m = 1; m < 16; m <<= 1)
#pragma unroll
    for (int g = 0; g < 8; ++g) wsum[g] += __shfl_xor(wsum[g], m, 32);
  float wb = ab2[0];
  if (colb == 0) {
#pragma unroll
    for (int g = 0; g < 8; ++g) sW[wave][g + hb * 8] = wsum[g] + wb;
  }
  LDS_FENCE();

  // ---- equivariant position accumulation ----
  if (lane < 16) {
    int e = e0 + lane;
    int vs = src[e], vd = dst[e];
    float w = sW[wave][lane];
    float rx = pos[vs * 3 + 0] - pos[vd * 3 + 0];
    float ry = pos[vs * 3 + 1] - pos[vd * 3 + 1];
    float rz = pos[vs * 3 + 2] - pos[vd * 3 + 2];
    float s = w * __frsqrt_rn(rx * rx + ry * ry + rz * rz);
    atomicAdd(&pos_acc[vd * 3 + 0], s * rx);
    atomicAdd(&pos_acc[vd * 3 + 1], s * ry);
    atomicAdd(&pos_acc[vd * 3 + 2], s * rz);
  }

  // ---- scatter m into intra (f32 atomics) ----
#pragma unroll
  for (int g = 0; g < 8; ++g) {
    int row = g + hb * 8;
    int vd = dst[e0 + row];
    float* base = intra + (size_t)vd * HD + colb;
#pragma unroll
    for (int nt = 0; nt < 8; ++nt) atomicAdd(base + nt * 16, mreg[nt][g]);
  }
}

// ---------------- node kernel: one wave per 16 nodes ----------------

__global__ __launch_bounds__(64) void egnn_node_kernel(
    const float* __restrict__ x, const float* __restrict__ intra,
    const float* __restrict__ deg, const float* __restrict__ pos_acc,
    float* __restrict__ pos,
    const unsigned short* __restrict__ nW1T, const float* __restrict__ nb1,
    const unsigned short* __restrict__ nW2T, const float* __restrict__ nb2,
    float* __restrict__ x_out, int N) {
  __shared__ unsigned short sA[2][16 * 256];
  __shared__ unsigned short sH[2][16 * 256];

  int wave = threadIdx.x >> 5, lane = threadIdx.x & 31;
  int tile = blockIdx.x * 2 + wave;
  if (tile * 16 >= N) return;
  int n0 = tile * 16;
  unsigned short* A = sA[wave];
  unsigned short* Hb = sH[wave];
  int hb = lane >> 4, colb = lane & 15;

  // ---- stage cat(x, intra/denom) -> bf16 [16,256] ----
  for (int r = 0; r < 16; ++r) {
    int v = n0 + r;
    float invd = 1.0f / fmaxf(deg[v], 1.0f);
    float4 fx = *(const float4*)(x + (size_t)v * HD + lane * 4);
    float4 fi = *(const float4*)(intra + (size_t)v * HD + lane * 4);
    int b0 = r * 256 + lane * 4;
    A[b0 + 0] = f2bf(fx.x); A[b0 + 1] = f2bf(fx.y); A[b0 + 2] = f2bf(fx.z); A[b0 + 3] = f2bf(fx.w);
    A[b0 + 128 + 0] = f2bf(fi.x * invd); A[b0 + 128 + 1] = f2bf(fi.y * invd);
    A[b0 + 128 + 2] = f2bf(fi.z * invd); A[b0 + 128 + 3] = f2bf(fi.w * invd);
  }
  // ---- pos += pos_acc / denom ----
  if (lane < 16) {
    int v = n0 + lane;
    float invd = 1.0f / fmaxf(deg[v], 1.0f);
#pragma unroll
    for (int c = 0; c < 3; ++c) pos[v * 3 + c] += pos_acc[v * 3 + c] * invd;
  }
  LDS_FENCE();

  // ---- GEMM1: h2 = relu([16,256] @ nW1 + nb1) -> bf16 LDS [16,256] ----
#pragma unroll
  for (int nt = 0; nt < 16; ++nt) {
    float bias = nb1[nt * 16 + colb];
    v8f c;
#pragma unroll
    for (int g = 0; g < 8; ++g) c[g] = bias;
#pragma unroll
    for (int kb = 0; kb < 8; ++kb)
      c = wmma_bf16(a_frag(A, 256, kb * 32, lane), b_frag(nW1T, 256, nt * 16, kb * 32, lane), c);
#pragma unroll
    for (int g = 0; g < 8; ++g) {
      int row = g + hb * 8;
      Hb[row * 256 + nt * 16 + colb] = f2bf(fmaxf(c[g], 0.0f));
    }
  }
  LDS_FENCE();

  // ---- GEMM2: x_out = h2 @ nW2 + nb2 -> f32 global [16,128] ----
#pragma unroll
  for (int nt = 0; nt < 8; ++nt) {
    float bias = nb2[nt * 16 + colb];
    v8f c;
#pragma unroll
    for (int g = 0; g < 8; ++g) c[g] = bias;
#pragma unroll
    for (int kb = 0; kb < 8; ++kb)
      c = wmma_bf16(a_frag(Hb, 256, kb * 32, lane), b_frag(nW2T, 256, nt * 16, kb * 32, lane), c);
#pragma unroll
    for (int g = 0; g < 8; ++g) {
      int row = g + hb * 8;
      x_out[(size_t)(n0 + row) * HD + nt * 16 + colb] = c[g];
    }
  }
}

// ---------------- host launcher ----------------

extern "C" void kernel_launch(void* const* d_in, const int* in_sizes, int n_in,
                              void* d_out, int out_size, void* d_ws, size_t ws_size,
                              hipStream_t stream) {
  (void)in_sizes; (void)n_in; (void)out_size; (void)ws_size;
  const float* x_in  = (const float*)d_in[0];
  const int*   ei    = (const int*)d_in[1];
  const float* eattr = (const float*)d_in[2];
  const float* posin = (const float*)d_in[3];
  const float* mW1 = (const float*)d_in[4];  const float* mb1 = (const float*)d_in[5];
  const float* mW2 = (const float*)d_in[6];  const float* mb2 = (const float*)d_in[7];
  const float* aW1 = (const float*)d_in[8];  const float* ab1 = (const float*)d_in[9];
  const float* aW2 = (const float*)d_in[10]; const float* ab2 = (const float*)d_in[11];
  const float* nW1 = (const float*)d_in[12]; const float* nb1 = (const float*)d_in[13];
  const float* nW2 = (const float*)d_in[14]; const float* nb2 = (const float*)d_in[15];

  char* wsb = (char*)d_ws;
  size_t off = 0;
  auto carve = [&](size_t bytes) -> void* {
    void* p = wsb + off;
    off += (bytes + 255) & ~(size_t)255;
    return p;
  };
  float* deg     = (float*)carve((size_t)NN * 4);
  float* pos_cur = (float*)carve((size_t)NN * 3 * 4);
  float* pos_acc = (float*)carve((size_t)NN * 3 * 4);
  float* intra   = (float*)carve((size_t)NN * HD * 4);
  float* xA      = (float*)carve((size_t)NN * HD * 4);
  float* xB      = (float*)carve((size_t)NN * HD * 4);
  unsigned short* mW1T = (unsigned short*)carve((size_t)3 * 384 * 128 * 2);
  unsigned short* mW2T = (unsigned short*)carve((size_t)3 * 128 * 128 * 2);
  unsigned short* aW1T = (unsigned short*)carve((size_t)3 * 128 * 256 * 2);
  unsigned short* nW1T = (unsigned short*)carve((size_t)3 * 256 * 256 * 2);
  unsigned short* nW2T = (unsigned short*)carve((size_t)3 * 256 * 128 * 2);

  hipMemsetAsync(deg, 0, (size_t)NN * 4, stream);
  hipMemcpyAsync(pos_cur, posin, (size_t)NN * 3 * 4, hipMemcpyDeviceToDevice, stream);
  egnn_deg_kernel<<<(NE + 255) / 256, 256, 0, stream>>>(ei + NE, deg, NE);

  for (int l = 0; l < 3; ++l) {
    egnn_wconv_kernel<<<(384 * 128 + 255) / 256, 256, 0, stream>>>(mW1 + (size_t)l * 384 * 128, mW1T + (size_t)l * 384 * 128, 384, 128);
    egnn_wconv_kernel<<<(128 * 128 + 255) / 256, 256, 0, stream>>>(mW2 + (size_t)l * 128 * 128, mW2T + (size_t)l * 128 * 128, 128, 128);
    egnn_wconv_kernel<<<(128 * 256 + 255) / 256, 256, 0, stream>>>(aW1 + (size_t)l * 128 * 256, aW1T + (size_t)l * 128 * 256, 128, 256);
    egnn_wconv_kernel<<<(256 * 256 + 255) / 256, 256, 0, stream>>>(nW1 + (size_t)l * 256 * 256, nW1T + (size_t)l * 256 * 256, 256, 256);
    egnn_wconv_kernel<<<(256 * 128 + 255) / 256, 256, 0, stream>>>(nW2 + (size_t)l * 256 * 128, nW2T + (size_t)l * 256 * 128, 256, 128);
  }

  const int edge_tiles = NE / 16;                 // 31250
  const int node_tiles = (NN + 15) / 16;          // 3125
  const int edge_grid = (edge_tiles + 1) / 2;     // 2 waves/block
  const int node_grid = (node_tiles + 1) / 2;

  for (int l = 0; l < 3; ++l) {
    const float* xl = (l == 0) ? x_in : ((l == 1) ? xA : xB);
    float* xn = (l == 1) ? xB : xA;
    hipMemsetAsync(intra, 0, (size_t)NN * HD * 4, stream);
    hipMemsetAsync(pos_acc, 0, (size_t)NN * 3 * 4, stream);
    egnn_edge_kernel<<<edge_grid, 64, 0, stream>>>(
        xl, eattr, ei, pos_cur,
        mW1T + (size_t)l * 384 * 128, mb1 + (size_t)l * 128,
        mW2T + (size_t)l * 128 * 128, mb2 + (size_t)l * 128,
        aW1T + (size_t)l * 128 * 256, ab1 + (size_t)l * 256,
        aW2 + (size_t)l * 256, ab2 + l,
        pos_acc, intra, NE);
    egnn_node_kernel<<<node_grid, 64, 0, stream>>>(
        xl, intra, deg, pos_acc, pos_cur,
        nW1T + (size_t)l * 256 * 256, nb1 + (size_t)l * 256,
        nW2T + (size_t)l * 256 * 128, nb2 + (size_t)l * 128,
        xn, NN);
  }

  hipMemcpyAsync(d_out, pos_cur, (size_t)NN * 3 * 4, hipMemcpyDeviceToDevice, stream);
}
